// MetaKernelDualAtt_48533130444997
// MI455X (gfx1250) — matrix operand order repaired and checked by reading, not compiled, not run on hardware
//
#include <hip/hip_runtime.h>
#include <hip/hip_bf16.h>
#include <math.h>

typedef __attribute__((ext_vector_type(16))) __bf16 v16bf;
typedef __attribute__((ext_vector_type(8)))  __bf16 v8bf;
typedef __attribute__((ext_vector_type(8)))  float  v8f;

#define B_    2
#define CIN_  64
#define COUT_ 64
#define H_    64
#define W_    2048
#define HW_   131072      // 2^17
#define K9_   576         // 9*CIN
#define EPS_  1e-5f
#define NTOT_ 262144.0f   // B*HW

// ---- LDS layout (byte offsets into one static block) ----
// B (bf16, wmma-swizzled)      : [0,      73728)
// union { pn4 f32 128*36           : 18432
//         A-chunks bf16 2x128*64   : 32768
//         out-stage f32 128*64     : 32768 } : [73728, 106496)
// s[128*9] f32                 : [106496, 111104)
// m[128*9] f32                 : [111104, 115712)
// wc[128*64] f32               : [115712, 148480)
// union { params 640 f32 / sumred 128 f32 } : [148480, 151040)
#define SMEM_TOTAL 151040

__global__ void __launch_bounds__(128) zero_sums_kernel(float* sums) {
    sums[threadIdx.x] = 0.0f;
}

__global__ void __launch_bounds__(256) fused_attn_gemm_kernel(
    const float* __restrict__ x, const int* __restrict__ mask,
    const float* __restrict__ w_spatial, const float* __restrict__ b_spatial,
    const float* __restrict__ w_channel, const float* __restrict__ b_channel,
    const float* __restrict__ w_agg, float* __restrict__ out,
    float* __restrict__ sums)
{
    __shared__ __align__(16) char smem[SMEM_TOTAL];
    __bf16* ldsBh  = (__bf16*)smem;
    float*  pnArr  = (float*)(smem + 73728);
    __bf16* ldsAh  = (__bf16*)(smem + 73728);   // two 8192-half buffers
    float*  stg    = (float*)(smem + 73728);
    float*  sArr   = (float*)(smem + 106496);
    float*  mArr   = (float*)(smem + 111104);
    float*  wcArr  = (float*)(smem + 115712);
    float*  prm    = (float*)(smem + 148480);
    float*  sumred = prm;

    const int tid = threadIdx.x;
    const int g0  = blockIdx.x * 128;          // global pixel base
    const int bb  = g0 >> 17;                  // batch (HW = 2^17)
    const int n0  = g0 & (HW_ - 1);
    const int y0  = n0 >> 11;                  // W = 2^11
    const int x0  = n0 & (W_ - 1);             // 128 px all in one row

    // ---------------- phase 0: stage small params + swizzle w_agg -> bf16 B frags ----
    for (int i = tid; i < 640; i += 256) {
        float v;
        if (i < 256)      v = w_spatial[i];
        else if (i < 320) v = b_spatial[i - 256];
        else if (i < 576) v = w_channel[i - 320];
        else              v = b_channel[i - 576];
        prm[i] = v;
    }
    {   // div-free B swizzle: thread t handles co = t&63, 144 contiguous k values
        int co = tid & 63;
        int kb = (tid >> 6) * 144;
        const float* wrow = w_agg + (size_t)co * K9_ + kb;
        int nt = co >> 4, nn = co & 15;
        for (int j = 0; j < 144; ++j) {
            int k = kb + j;
            int ktile = k >> 5, kk = k & 31;
            int lane = (kk < 16) ? nn : (nn + 16);
            ldsBh[(size_t)(ktile * 4 + nt) * 512 + lane * 16 + (kk & 15)] = (__bf16)wrow[j];
        }
    }

    // ---------------- phase 1a: stage pn4 + mask for my pixel (2 threads/pixel) -----
    const int px = tid >> 1;
    const int hf = tid & 1;
    const int xx = x0 + px;
    {
        int kbeg = hf ? 4 : 0, kend = hf ? 9 : 5;
        for (int k = kbeg; k < kend; ++k) {
            int dy = k / 3 - 1, dx = k % 3 - 1;
            int yy = y0 + dy, xc = xx + dx;
            bool v = (yy >= 0) && (yy < H_) && (xc >= 0) && (xc < W_);
            float p0 = 0.f, p1 = 0.f, p2 = 0.f, mk = 0.f;
            if (v) {
                size_t o = (size_t)yy * W_ + xc;
                size_t pb = (size_t)bb * 67 * HW_;
                p0 = x[pb + o];
                p1 = x[pb + HW_ + o];
                p2 = x[pb + 2 * (size_t)HW_ + o];
                mk = (float)mask[(size_t)bb * HW_ + o];
            }
            float r = sqrtf(p0 * p0 + p1 * p1 + p2 * p2);
            float* pn = pnArr + (px * 9 + k) * 4;
            pn[0] = p0; pn[1] = p1; pn[2] = p2; pn[3] = r;
            mArr[px * 9 + k] = mk;
        }
    }
    __syncthreads();

    // ---------------- phase 1b: attention weights ----------------------------------
    {
        const float* pnp = pnArr + px * 36;
        float c0 = pnp[16], c1 = pnp[17], c2 = pnp[18], c3 = pnp[19];
        float P[9][4];
#pragma unroll
        for (int k = 0; k < 9; ++k) {
            P[k][0] = pnp[4 * k + 0] - c0; P[k][1] = pnp[4 * k + 1] - c1;
            P[k][2] = pnp[4 * k + 2] - c2; P[k][3] = pnp[4 * k + 3] - c3;
        }
        float wsk[9];
#pragma unroll
        for (int k = 0; k < 9; ++k) {
            float mx = -3.4e38f;
            for (int c = 0; c < 32; ++c) {
                int cc = hf * 32 + c;
                float d = prm[256 + cc];
                d = fmaf(P[k][0], prm[cc * 4 + 0], d);
                d = fmaf(P[k][1], prm[cc * 4 + 1], d);
                d = fmaf(P[k][2], prm[cc * 4 + 2], d);
                d = fmaf(P[k][3], prm[cc * 4 + 3], d);
                mx = fmaxf(mx, d);
            }
            wsk[k] = mx;
        }
#pragma unroll
        for (int k = 0; k < 9; ++k) wsk[k] = fmaxf(wsk[k], __shfl_xor(wsk[k], 1));
        float wcl[32];
#pragma unroll
        for (int c = 0; c < 32; ++c) {
            int cc = hf * 32 + c;
            float w0 = prm[320 + cc * 4 + 0], w1 = prm[320 + cc * 4 + 1];
            float w2 = prm[320 + cc * 4 + 2], w3 = prm[320 + cc * 4 + 3];
            float bc = prm[576 + cc];
            float mx = -3.4e38f;
#pragma unroll
            for (int k = 0; k < 9; ++k) {
                float d = fmaf(P[k][0], w0, fmaf(P[k][1], w1,
                          fmaf(P[k][2], w2, fmaf(P[k][3], w3, bc))));
                mx = fmaxf(mx, d);
            }
            wcl[c] = mx;
        }
        float mxs = wsk[0];
#pragma unroll
        for (int k = 1; k < 9; ++k) mxs = fmaxf(mxs, wsk[k]);
        float es[9]; float ssum = 0.f;
#pragma unroll
        for (int k = 0; k < 9; ++k) { es[k] = __expf(wsk[k] - mxs); ssum += es[k]; }
        float invs = 1.0f / ssum;
        if (!hf) {
#pragma unroll
            for (int k = 0; k < 9; ++k) sArr[px * 9 + k] = es[k] * invs * mArr[px * 9 + k];
        }
        float lmx = wcl[0];
#pragma unroll
        for (int c = 1; c < 32; ++c) lmx = fmaxf(lmx, wcl[c]);
        float gmx = fmaxf(lmx, __shfl_xor(lmx, 1));
        float ls = 0.f;
#pragma unroll
        for (int c = 0; c < 32; ++c) { wcl[c] = __expf(wcl[c] - gmx); ls += wcl[c]; }
        float gs = ls + __shfl_xor(ls, 1);
        float invc = 1.0f / gs;
#pragma unroll
        for (int c = 0; c < 32; ++c) wcArr[px * 64 + hf * 32 + c] = wcl[c] * invc;
    }
    __syncthreads();   // phase1 writes -> phase2 reads

    // ---------------- phase 2: double-buffered, register-staged bf16 WMMA GEMM -----
    const int wave = tid >> 5, lane = tid & 31;
    const int pixbase = wave * 16;
    const int mrow = lane & 15;
    const int hsel = (lane & 16) ? 16 : 0;
    const int p  = tid & 127;          // fill pixel (fixed per thread)
    const int cg = (tid >> 7) * 32;    // fill ci base (0 or 32)

    // wc[ci] invariant across neighbors: hoist to registers
    float wcv[32];
#pragma unroll
    for (int i = 0; i < 32; ++i) wcv[i] = wcArr[p * 64 + cg + i];

    const float* fbase = x + ((size_t)(bb * 67 + 3 + cg)) * HW_;
    float fregs[32];

    // prologue: load + weight + store neighbor k=0 into buffer 0
    {
        int yy = y0 - 1, xc = x0 + p - 1;
        bool ok = (yy >= 0) && (xc >= 0) && (xc < W_);
        if (ok) {
            const float* src = fbase + (size_t)yy * W_ + xc;
#pragma unroll
            for (int i = 0; i < 32; ++i) fregs[i] = src[(size_t)i * HW_];
        } else {
#pragma unroll
            for (int i = 0; i < 32; ++i) fregs[i] = 0.f;
        }
        float sk = sArr[p * 9 + 0], mk = mArr[p * 9 + 0];
        __bf16 tmp[32];
#pragma unroll
        for (int i = 0; i < 32; ++i) tmp[i] = (__bf16)(fregs[i] * (sk + wcv[i] * mk));
        v8bf* dst = (v8bf*)(ldsAh + p * 64 + cg);
#pragma unroll
        for (int j = 0; j < 4; ++j) dst[j] = *(const v8bf*)(tmp + 8 * j);
    }

    v8f acc[4] = {};
    for (int k = 0; k < 9; ++k) {
        __syncthreads();   // A buffer (k&1) filled; reads of other buffer done
        // stage next neighbor's features in registers (overlaps with WMMA below)
        int kn = k + 1;
        bool okn = false;
        if (kn < 9) {
            int yy = y0 + (kn / 3 - 1);
            int xc = x0 + p + (kn % 3 - 1);
            okn = (yy >= 0) && (yy < H_) && (xc >= 0) && (xc < W_);
            if (okn) {
                const float* src = fbase + (size_t)yy * W_ + xc;
#pragma unroll
                for (int i = 0; i < 32; ++i) fregs[i] = src[(size_t)i * HW_];
            }
        }
        if (k < 7) {   // prefetch neighbor k+2 slab (speculative)
            int k2 = k + 2;
            int yy2 = y0 + (k2 / 3 - 1);
            int xc2 = x0 + p + (k2 % 3 - 1);
            if (yy2 >= 0 && yy2 < H_ && xc2 >= 0 && xc2 < W_)
                __builtin_prefetch((const void*)(fbase + (size_t)yy2 * W_ + xc2), 0, 1);
        }
        // WMMA on current buffer
        const int abase = (k & 1) * 8192;
#pragma unroll
        for (int kt = 0; kt < 2; ++kt) {
            int pix = pixbase + mrow;
            int base0 = abase + pix * 64 + kt * 32 + hsel;
            v8bf alo = *(const v8bf*)(ldsAh + base0);
            v8bf ahi = *(const v8bf*)(ldsAh + base0 + 8);
            v16bf af;
#pragma unroll
            for (int i = 0; i < 8; ++i) { af[i] = alo[i]; af[i + 8] = ahi[i]; }
            int kg = k * 2 + kt;
            const __bf16* bp = ldsBh + (size_t)(kg * 4) * 512 + lane * 16;
            v16bf b0 = *(const v16bf*)(bp);
            v16bf b1 = *(const v16bf*)(bp + 512);
            v16bf b2 = *(const v16bf*)(bp + 1024);
            v16bf b3 = *(const v16bf*)(bp + 1536);
            acc[0] = __builtin_amdgcn_wmma_f32_16x16x32_bf16(false, af, false, b0, (short)0, acc[0], false, false);
            acc[1] = __builtin_amdgcn_wmma_f32_16x16x32_bf16(false, af, false, b1, (short)0, acc[1], false, false);
            acc[2] = __builtin_amdgcn_wmma_f32_16x16x32_bf16(false, af, false, b2, (short)0, acc[2], false, false);
            acc[3] = __builtin_amdgcn_wmma_f32_16x16x32_bf16(false, af, false, b3, (short)0, acc[3], false, false);
        }
        // weight + convert + store next buffer
        if (kn < 9) {
            float sk = sArr[p * 9 + kn], mk = mArr[p * 9 + kn];
            __bf16 tmp[32];
#pragma unroll
            for (int i = 0; i < 32; ++i) {
                float f = okn ? fregs[i] : 0.f;
                tmp[i] = (__bf16)(f * (sk + wcv[i] * mk));
            }
            v8bf* dst = (v8bf*)(ldsAh + ((kn & 1) * 8192) + p * 64 + cg);
#pragma unroll
            for (int j = 0; j < 4; ++j) dst[j] = *(const v8bf*)(tmp + 8 * j);
        }
    }

    // ---------------- phase 3: stage, coalesced store, channel sums ----------------
    __syncthreads();   // stage overlaps A buffers
#pragma unroll
    for (int nt = 0; nt < 4; ++nt) {
#pragma unroll
        for (int j = 0; j < 8; ++j) {
            int pix = pixbase + j + ((lane & 16) ? 8 : 0);
            int co = nt * 16 + (lane & 15);
            stg[pix * 64 + co] = acc[nt][j];
        }
    }
    if (tid < 128) sumred[tid] = 0.f;
    __syncthreads();
    for (int idx = tid; idx < 8192; idx += 256) {
        int co = idx >> 7, q = idx & 127;
        out[((size_t)(bb * COUT_ + co)) * HW_ + n0 + q] = stg[q * 64 + co];
    }
    {
        int co = tid & 63, g = tid >> 6;
        float s1 = 0.f, s2 = 0.f;
        for (int q = g * 32; q < g * 32 + 32; ++q) {
            float v = stg[q * 64 + co];
            s1 += v; s2 += v * v;
        }
        atomicAdd(&sumred[co], s1);
        atomicAdd(&sumred[64 + co], s2);
    }
    __syncthreads();
    if (tid < 128) atomicAdd(&sums[tid], sumred[tid]);
}

__global__ void __launch_bounds__(256) norm_relu_kernel(
    float* __restrict__ out, const float* __restrict__ sums,
    const float* __restrict__ gamma, const float* __restrict__ beta)
{
    size_t i = ((size_t)blockIdx.x * 256 + threadIdx.x) * 4;
    int co = (int)((i >> 17) & 63);
    const float invn = 1.0f / NTOT_;
    float mean = sums[co] * invn;
    float var  = sums[64 + co] * invn - mean * mean;
    float sc = rsqrtf(var + EPS_) * gamma[co];
    float sh = beta[co] - mean * sc;
    float4 v = *(const float4*)(out + i);
    v.x = fmaxf(fmaf(v.x, sc, sh), 0.f);
    v.y = fmaxf(fmaf(v.y, sc, sh), 0.f);
    v.z = fmaxf(fmaf(v.z, sc, sh), 0.f);
    v.w = fmaxf(fmaf(v.w, sc, sh), 0.f);
    *(float4*)(out + i) = v;
}

extern "C" void kernel_launch(void* const* d_in, const int* in_sizes, int n_in,
                              void* d_out, int out_size, void* d_ws, size_t ws_size,
                              hipStream_t stream) {
    (void)in_sizes; (void)n_in; (void)out_size; (void)ws_size;
    const float* x         = (const float*)d_in[0];
    const int*   mask      = (const int*)d_in[1];
    const float* w_spatial = (const float*)d_in[2];
    const float* b_spatial = (const float*)d_in[3];
    const float* w_channel = (const float*)d_in[4];
    const float* b_channel = (const float*)d_in[5];
    const float* w_agg     = (const float*)d_in[6];
    const float* gamma     = (const float*)d_in[7];
    const float* beta      = (const float*)d_in[8];
    float* out  = (float*)d_out;
    float* sums = (float*)d_ws;

    zero_sums_kernel<<<1, 128, 0, stream>>>(sums);
    fused_attn_gemm_kernel<<<(B_ * HW_) / 128, 256, 0, stream>>>(
        x, mask, w_spatial, b_spatial, w_channel, b_channel, w_agg, out, sums);
    norm_relu_kernel<<<(B_ * COUT_ * HW_) / 1024, 256, 0, stream>>>(out, sums, gamma, beta);
}
